// SparsePagedAttention_90787018703115
// MI455X (gfx1250) — matrix operand order repaired and checked by reading, not compiled, not run on hardware
//
#include <hip/hip_runtime.h>
#include <hip/hip_bf16.h>

// ---------------- problem constants ----------------
#define B_   2
#define S_   2048
#define H_   32
#define HK_  8
#define D_   128
#define G_   (H_ / HK_)           // 4
#define SCALE_ 0.08838834764831845f   // 1/sqrt(128)
#define THRESH_ 0.99f
#define BUF_ 128
#define NEG_BIG (-3.402823466e38f)

typedef __attribute__((ext_vector_type(16))) _Float16 v16h;
typedef __attribute__((ext_vector_type(8)))  float    v8f;

__device__ __forceinline__ float half_red_max(float x) {
    x = fmaxf(x, __shfl_xor(x, 1, 16));
    x = fmaxf(x, __shfl_xor(x, 2, 16));
    x = fmaxf(x, __shfl_xor(x, 4, 16));
    x = fmaxf(x, __shfl_xor(x, 8, 16));
    return x;
}
__device__ __forceinline__ float half_red_sum(float x) {
    x += __shfl_xor(x, 1, 16);
    x += __shfl_xor(x, 2, 16);
    x += __shfl_xor(x, 4, 16);
    x += __shfl_xor(x, 8, 16);
    return x;
}

// ---------------- phase 0: mean of V per (b,hk) ----------------
__global__ __launch_bounds__(D_)
void meanv_kernel(const float* __restrict__ v, float* __restrict__ meanV) {
    int b = blockIdx.x / HK_, hk = blockIdx.x % HK_, d = threadIdx.x;
    float s = 0.f;
    const float* vp = v + ((size_t)b * S_ * HK_ + hk) * D_ + d;
    for (int t = 0; t < S_; ++t) s += vp[(size_t)t * HK_ * D_];
    meanV[((size_t)b * HK_ + hk) * D_ + d] = s * (1.0f / S_);
}

// ---------------- phase 1: flash attention fwd (one wave per 16 q-rows) ----------------
__global__ __launch_bounds__(32)
void attn_fwd_kernel(const float* __restrict__ q, const float* __restrict__ k,
                     const float* __restrict__ v, const int* __restrict__ plens,
                     float* __restrict__ out,
                     float* __restrict__ Mrow, float* __restrict__ Lrow) {
    const int QT = S_ / 16;
    int wid = blockIdx.x;
    int qt = wid % QT, h = (wid / QT) % H_, b = wid / (QT * H_);
    int q0 = qt * 16, hk = h / G_;
    int lane = threadIdx.x, n16 = lane & 15, sel = lane >> 4;

    __shared__ _Float16 psh[16][32];

    // Q A-fragments (16x32 f16 layout): K=16*sel+e (contiguous)
    v16h Aq[4];
    {
        const float* qp = q + (((size_t)b * S_ + q0 + n16) * H_ + h) * D_;
        #pragma unroll
        for (int f = 0; f < 4; ++f) {
            #pragma unroll
            for (int e = 0; e < 8; ++e) {
                Aq[f][e]     = (_Float16)qp[32 * f + 16 * sel + e];
                Aq[f][8 + e] = (_Float16)qp[32 * f + 16 * sel + 8 + e];
            }
        }
    }

    float mrow[8], lrow[8];
    v8f O[8];
    #pragma unroll
    for (int p = 0; p < 8; ++p) { mrow[p] = NEG_BIG; lrow[p] = 0.f; }
    #pragma unroll
    for (int dt = 0; dt < 8; ++dt)
        #pragma unroll
        for (int p = 0; p < 8; ++p) O[dt][p] = 0.f;

    int jbmax = (q0 + 15) / 32;           // t0 = 32*jb <= q0 always
    for (int jb = 0; jb <= jbmax; ++jb) {
        int t0 = jb * 32;
        // ---- scores: two 16-key n-tiles ----
        v8f sc[2];
        #pragma unroll
        for (int nt = 0; nt < 2; ++nt) {
            v8f acc;
            #pragma unroll
            for (int p = 0; p < 8; ++p) acc[p] = 0.f;
            int tok = t0 + nt * 16 + n16;
            const float* kp = k + (((size_t)b * S_ + tok) * HK_ + hk) * D_ + 16 * sel;
            #pragma unroll
            for (int f = 0; f < 4; ++f) {
                v16h bk;  // B 32x16: K = 16*sel + e (contiguous in d)
                #pragma unroll
                for (int e = 0; e < 16; ++e) bk[e] = (_Float16)kp[32 * f + e];
                acc = __builtin_amdgcn_wmma_f32_16x16x32_f16(false, Aq[f], false, bk,
                                                             (short)0, acc, false, false);
            }
            sc[nt] = acc;
        }
        // ---- scale + causal mask + online softmax ----
        float mnew[8], alpha[8];
        #pragma unroll
        for (int p = 0; p < 8; ++p) {
            int qrow = q0 + p + 8 * sel;
            float s0 = sc[0][p] * SCALE_;
            float s1 = sc[1][p] * SCALE_;
            if (t0 + n16 > qrow)      s0 = NEG_BIG;
            if (t0 + 16 + n16 > qrow) s1 = NEG_BIG;
            sc[0][p] = s0; sc[1][p] = s1;
            float vm = half_red_max(fmaxf(s0, s1));
            mnew[p]  = fmaxf(mrow[p], vm);
            alpha[p] = __expf(mrow[p] - mnew[p]);
        }
        #pragma unroll
        for (int p = 0; p < 8; ++p) {
            float p0 = __expf(sc[0][p] - mnew[p]);
            float p1 = __expf(sc[1][p] - mnew[p]);
            sc[0][p] = p0; sc[1][p] = p1;
            float rs = half_red_sum(p0 + p1);
            lrow[p] = lrow[p] * alpha[p] + rs;
            mrow[p] = mnew[p];
            #pragma unroll
            for (int dt = 0; dt < 8; ++dt) O[dt][p] *= alpha[p];
        }
        // ---- stage P (D-layout) through LDS into A-layout ----
        __syncthreads();
        #pragma unroll
        for (int p = 0; p < 8; ++p) {
            psh[p + 8 * sel][n16]      = (_Float16)sc[0][p];
            psh[p + 8 * sel][16 + n16] = (_Float16)sc[1][p];
        }
        __syncthreads();
        v16h ap;
        #pragma unroll
        for (int e = 0; e < 8; ++e) {
            ap[e]     = psh[n16][16 * sel + e];
            ap[8 + e] = psh[n16][16 * sel + 8 + e];
        }
        // ---- P·V ----
        const float* vp = v + (((size_t)b * S_ + t0 + 16 * sel) * HK_ + hk) * D_ + n16;
        #pragma unroll
        for (int dt = 0; dt < 8; ++dt) {
            v16h bv;  // B 32x16: token = t0 + 16*sel + e, col = dt*16 + n16
            #pragma unroll
            for (int e = 0; e < 16; ++e)
                bv[e] = (_Float16)vp[(size_t)e * HK_ * D_ + dt * 16];
            O[dt] = __builtin_amdgcn_wmma_f32_16x16x32_f16(false, ap, false, bv,
                                                           (short)0, O[dt], false, false);
        }
    }
    // ---- finalize ----
    #pragma unroll
    for (int p = 0; p < 8; ++p) {
        float inv = 1.0f / lrow[p];
        int qrow = q0 + p + 8 * sel;
        float* op = out + (((size_t)b * S_ + qrow) * H_ + h) * D_;
        #pragma unroll
        for (int dt = 0; dt < 8; ++dt) op[dt * 16 + n16] = O[dt][p] * inv;
        if (n16 == 0) {
            Mrow[((size_t)b * H_ + h) * S_ + qrow] = mrow[p];
            Lrow[((size_t)b * H_ + h) * S_ + qrow] = lrow[p];
        }
    }
    (void)plens;
}

// ---------------- phase 1c: rows q >= prompt_len become mean(V) ----------------
__global__ __launch_bounds__(256)
void fillmean_kernel(const int* __restrict__ plens, const float* __restrict__ meanV,
                     float* __restrict__ out) {
    int qq = blockIdx.x % S_, b = blockIdx.x / S_;
    if (qq < plens[b]) return;
    for (int i = threadIdx.x; i < H_ * D_; i += 256) {
        int h = i / D_, d = i % D_;
        out[((size_t)b * S_ + qq) * (H_ * D_) + i] =
            meanV[((size_t)b * HK_ + h / G_) * D_ + d];
    }
}

// ---------------- phase 2: per-kv-token attention mass (recompute pass) ----------------
__global__ __launch_bounds__(32)
void mass_kernel(const float* __restrict__ q, const float* __restrict__ k,
                 const int* __restrict__ plens,
                 const float* __restrict__ Mrow, const float* __restrict__ Lrow,
                 float* __restrict__ sArr) {
    const int TT = S_ / 16;
    int wid = blockIdx.x;
    int tt = wid % TT, hk = (wid / TT) % HK_, b = wid / (TT * HK_);
    int t0 = tt * 16;
    int lane = threadIdx.x, n16 = lane & 15, sel = lane >> 4;
    int plen = plens[b];

    float accs[8];
    #pragma unroll
    for (int p = 0; p < 8; ++p) accs[p] = 0.f;

    // A fragments from K tile: row m = token t0+n16
    v16h Ak[4];
    {
        const float* kp = k + (((size_t)b * S_ + t0 + n16) * HK_ + hk) * D_;
        #pragma unroll
        for (int f = 0; f < 4; ++f) {
            #pragma unroll
            for (int e = 0; e < 8; ++e) {
                Ak[f][e]     = (_Float16)kp[32 * f + 16 * sel + e];
                Ak[f][8 + e] = (_Float16)kp[32 * f + 16 * sel + 8 + e];
            }
        }
    }

    int jq0 = t0 / 16;
    int jq1 = (plen + 15) / 16;   // exclusive
    for (int g = 0; g < G_; ++g) {
        int h = hk * G_ + g;
        for (int jq = jq0; jq < jq1; ++jq) {
            int qcol = jq * 16 + n16;
            v8f acc;
            #pragma unroll
            for (int p = 0; p < 8; ++p) acc[p] = 0.f;
            const float* qp = q + (((size_t)b * S_ + qcol) * H_ + h) * D_ + 16 * sel;
            #pragma unroll
            for (int f = 0; f < 4; ++f) {
                v16h bq;  // B 32x16: d = 32f + 16*sel + e, col = q
                #pragma unroll
                for (int e = 0; e < 16; ++e) bq[e] = (_Float16)qp[32 * f + e];
                acc = __builtin_amdgcn_wmma_f32_16x16x32_f16(false, Ak[f], false, bq,
                                                             (short)0, acc, false, false);
            }
            float Mq = Mrow[((size_t)b * H_ + h) * S_ + qcol];
            float Lq = Lrow[((size_t)b * H_ + h) * S_ + qcol];
            float invL = 1.0f / Lq;
            bool qvalid = (qcol < plen);
            #pragma unroll
            for (int p = 0; p < 8; ++p) {
                int tok = t0 + p + 8 * sel;
                if (qvalid && tok <= qcol)
                    accs[p] += __expf(acc[p] * SCALE_ - Mq) * invL;
            }
        }
    }
    #pragma unroll
    for (int p = 0; p < 8; ++p) {
        float s = half_red_sum(accs[p]);
        int tok = t0 + p + 8 * sel;
        if (n16 == 0)
            sArr[((size_t)b * HK_ + hk) * S_ + tok] = (tok < plen) ? s : 0.f;
    }
}

// ---------------- phase 3: sort + cumsum + count per (b,hk) ----------------
__global__ __launch_bounds__(256)
void prune_kernel(const float* __restrict__ sArr, const int* __restrict__ plens,
                  int* __restrict__ outTail) {
    __shared__ float a[S_];
    __shared__ float part[257];
    __shared__ int cnt;
    int pair = blockIdx.x;              // pair = b*HK + hk
    int hk = pair % HK_, b = pair / HK_;
    int tid = threadIdx.x;
    for (int i = tid; i < S_; i += 256)
        a[i] = sArr[((size_t)b * HK_ + hk) * S_ + i];
    if (tid == 0) cnt = 0;
    __syncthreads();
    // bitonic sort, descending
    for (int kk = 2; kk <= S_; kk <<= 1) {
        for (int j = kk >> 1; j > 0; j >>= 1) {
            for (int i = tid; i < S_; i += 256) {
                int ix = i ^ j;
                if (ix > i) {
                    float x = a[i], y = a[ix];
                    bool desc = ((i & kk) == 0);
                    bool sw = desc ? (x < y) : (x > y);
                    if (sw) { a[i] = y; a[ix] = x; }
                }
            }
            __syncthreads();
        }
    }
    // block-wide inclusive scan over 2048 sorted values (chunks of 8)
    float loc[8], csum = 0.f;
    #pragma unroll
    for (int e = 0; e < 8; ++e) { loc[e] = a[tid * 8 + e]; csum += loc[e]; }
    part[tid] = csum;
    __syncthreads();
    if (tid == 0) {
        float r = 0.f;
        for (int i = 0; i < 256; ++i) { float t = part[i]; part[i] = r; r += t; }
        part[256] = r;
    }
    __syncthreads();
    float total = part[256];
    float T = THRESH_ * total;
    float run = part[tid];
    int c = 0;
    #pragma unroll
    for (int e = 0; e < 8; ++e) { run += loc[e]; if (run < T) c++; }
    atomicAdd(&cnt, c);
    __syncthreads();
    if (tid == 0) {
        int count = cnt + 1;
        int plen = plens[b];
        int saved = count < BUF_ ? BUF_ : count;
        if (saved > plen) saved = plen;
        outTail[pair] = saved;
    }
}

// ---------------- launch ----------------
extern "C" void kernel_launch(void* const* d_in, const int* in_sizes, int n_in,
                              void* d_out, int out_size, void* d_ws, size_t ws_size,
                              hipStream_t stream) {
    (void)in_sizes; (void)n_in; (void)out_size; (void)ws_size;
    const float* q     = (const float*)d_in[0];
    const float* k     = (const float*)d_in[1];
    const float* v     = (const float*)d_in[2];
    const int*   plens = (const int*)d_in[3];

    float* out = (float*)d_out;
    int* tail  = (int*)((float*)d_out + (size_t)B_ * S_ * H_ * D_);

    float* wsf   = (float*)d_ws;
    float* Mrow  = wsf;                                  // B*H*S
    float* Lrow  = Mrow + (size_t)B_ * H_ * S_;          // B*H*S
    float* meanV = Lrow + (size_t)B_ * H_ * S_;          // B*HK*D
    float* sArr  = meanV + (size_t)B_ * HK_ * D_;        // B*HK*S

    meanv_kernel<<<B_ * HK_, D_, 0, stream>>>(v, meanV);
    attn_fwd_kernel<<<B_ * H_ * (S_ / 16), 32, 0, stream>>>(q, k, v, plens, out, Mrow, Lrow);
    fillmean_kernel<<<B_ * S_, 256, 0, stream>>>(plens, meanV, out);
    mass_kernel<<<B_ * HK_ * (S_ / 16), 32, 0, stream>>>(q, k, plens, Mrow, Lrow, sArr);
    prune_kernel<<<B_ * HK_, 256, 0, stream>>>(sArr, plens, tail);
}
